// SphericalUNet_41695542509837
// MI455X (gfx1250) — matrix operand order repaired and checked by reading, not compiled, not run on hardware
//
#include <hip/hip_runtime.h>

#define NBATCH 16
#define EPSV   1e-5f

typedef __attribute__((ext_vector_type(16))) __bf16          v16bf;
typedef __attribute__((ext_vector_type(8)))  unsigned int    v8ui;
typedef __attribute__((ext_vector_type(8)))  float           v8f;

__device__ __forceinline__ unsigned short f2bf(float f) {
    unsigned u = __builtin_bit_cast(unsigned, f);
    u += 0x7FFFu + ((u >> 16) & 1u);          // round-to-nearest-even
    return (unsigned short)(u >> 16);
}

// ---------------------------------------------------------------------------
// Pack W[O,C,KN] (f32) into bf16 A-fragment order:
//   Wp[(((ot*KN + k)*nc + cch)*32 + lane)*8 + i] = {bf16(ka0), bf16(ka0+1)}
// so a lane loads its entire 16-bf16 A fragment as one contiguous 32B vector.
// ---------------------------------------------------------------------------
__global__ __launch_bounds__(256)
void k_pack_w(const float* __restrict__ W, unsigned* __restrict__ Wp,
              int O, int C, int KN, int otiles, int nc)
{
    int gid = blockIdx.x * 256 + threadIdx.x;
    int total = otiles * KN * nc * 256;
    if (gid >= total) return;
    int i    = gid & 7;
    int lane = (gid >> 3) & 31;
    int rest = gid >> 8;
    int cch  = rest % nc;  rest /= nc;
    int k    = rest % KN;
    int ot   = rest / KN;
    int half = lane >> 4, sub = lane & 15;
    int row  = ot * 16 + sub;
    int ka0  = ((i < 4) ? (2*i) : (16 + 2*(i-4))) + 8*half;
    int c0   = cch * 32 + ka0;
    float w0 = (row < O && c0     < C) ? W[((size_t)row * C + c0    ) * KN + k] : 0.f;
    float w1 = (row < O && c0 + 1 < C) ? W[((size_t)row * C + c0 + 1) * KN + k] : 0.f;
    Wp[gid] = (unsigned)f2bf(w0) | ((unsigned)f2bf(w1) << 16);
}

__global__ void k_zero(float* p, int n)
{
    int i = blockIdx.x * 64 + threadIdx.x;
    if (i < n) p[i] = 0.f;
}

// ---------------------------------------------------------------------------
// Unified conv kernel:  Y[b,o,v] = bias[o] + sum_{k<KN} sum_{c<C} W[o,c,k]*X[b,c,col(v,k)]
//   col(v,k) = nbr ? nbr[v*KN+k] : v     (KN==1 -> conv1x1, KN==7 -> meshconv)
// Block = 4 waves x 32 output columns. The 32x32 f32 X sub-tile is gathered
// into LDS with GLOBAL_LOAD_ASYNC_TO_LDS_B32 (ASYNCcnt), then each wave builds
// two bf16 B fragments (converting in registers) and reuses its pre-packed A
// fragment across two v_wmma_f32_16x16x32_bf16 issues.
// ---------------------------------------------------------------------------
__global__ __launch_bounds__(128)
void k_conv_wmma(const float* __restrict__ X, const unsigned* __restrict__ Wp,
                 const float* __restrict__ bias, const int* __restrict__ nbr,
                 const float* __restrict__ zbuf,
                 float* __restrict__ Y, int C, int Vin, int O, int Vout, int KN,
                 int otiles, int nc)
{
    __shared__ int   scol[32];
    __shared__ float sX[32 * 32];         // [k row 0..31][col n 0..31]

    const int tid  = threadIdx.x;
    const int lane = tid & 31;
    const int wave = tid >> 5;
    const int half = lane >> 4, sub = lane & 15;
    const int v0 = blockIdx.x * 32;
    const int b  = blockIdx.z;

    int ot = blockIdx.y * 4 + wave;
    if (ot >= otiles) ot = otiles - 1;    // clamp: duplicate work, uniform flow
    const int o0 = ot * 16;

    const float* Xb = X + (size_t)b * C * Vin;

    v8f acc0 = {0.f,0.f,0.f,0.f,0.f,0.f,0.f,0.f};
    v8f acc1 = {0.f,0.f,0.f,0.f,0.f,0.f,0.f,0.f};

    for (int k = 0; k < KN; ++k) {
        __syncthreads();
        if (tid < 32) {
            int vv = v0 + tid;
            scol[tid] = (vv < Vout) ? (nbr ? nbr[vv * KN + k] : vv) : 0;
        }
        __syncthreads();
        for (int c0 = 0; c0 < C; c0 += 32) {
            // async-gather the 32x32 f32 tile straight into LDS (no VGPR data)
            for (int e = tid; e < 1024; e += 128) {
                int kk = e >> 5, n = e & 31;
                int cc = c0 + kk;
                const float* src = (cc < C) ? (Xb + (size_t)cc * Vin + scol[n])
                                            : (zbuf + n);
                unsigned           laddr = (unsigned)(size_t)(&sX[e]);
                unsigned long long gaddr = (unsigned long long)(size_t)src;
                asm volatile("global_load_async_to_lds_b32 %0, %1, off"
                             :: "v"(laddr), "v"(gaddr) : "memory");
            }
            asm volatile("s_wait_asynccnt 0x0" ::: "memory");
            __syncthreads();
            // A fragment: one contiguous 32B vector per lane from packed weights
            v8ui aw = *(const v8ui*)(Wp + ((((size_t)ot * KN + k) * nc + (c0 >> 5)) * 32 + lane) * 8);
            v16bf af = __builtin_bit_cast(v16bf, aw);
            // Two B fragments (cols sub and sub+16), bf16-convert from LDS
            v8ui bw0, bw1;
            #pragma unroll
            for (int i = 0; i < 8; ++i) {
                int kb0 = ((i < 4) ? (2 * i) : (16 + 2 * (i - 4))) + 8 * half;
                int n0  = sub;
                int n1  = sub + 16;
                bw0[i] = (unsigned)f2bf(sX[kb0 * 32 + n0]) |
                         ((unsigned)f2bf(sX[(kb0 + 1) * 32 + n0]) << 16);
                bw1[i] = (unsigned)f2bf(sX[kb0 * 32 + n1]) |
                         ((unsigned)f2bf(sX[(kb0 + 1) * 32 + n1]) << 16);
            }
            v16bf bf0 = __builtin_bit_cast(v16bf, bw0);
            v16bf bf1 = __builtin_bit_cast(v16bf, bw1);
            acc0 = __builtin_amdgcn_wmma_f32_16x16x32_bf16(false, af, false, bf0,
                                                           (short)0, acc0, false, false);
            acc1 = __builtin_amdgcn_wmma_f32_16x16x32_bf16(false, af, false, bf1,
                                                           (short)0, acc1, false, false);
            __syncthreads();
        }
    }
    // C/D: vgpr r -> M = r + 8*half, N = sub (+16 for second tile)
    int vc0 = v0 + sub, vc1 = v0 + 16 + sub;
    #pragma unroll
    for (int r = 0; r < 8; ++r) {
        int oo = o0 + r + 8 * half;
        if (oo < O) {
            float bs = bias[oo];
            if (vc0 < Vout) Y[((size_t)b * O + oo) * Vout + vc0] = acc0[r] + bs;
            if (vc1 < Vout) Y[((size_t)b * O + oo) * Vout + vc1] = acc1[r] + bs;
        }
    }
}

// ---------------------------------------------------------------------------
// BatchNorm statistics: one block per channel, reduce over (B,V)
// ---------------------------------------------------------------------------
__global__ __launch_bounds__(256)
void k_bn_stats(const float* __restrict__ x, int C, int V,
                float* __restrict__ mean, float* __restrict__ rstd)
{
    const int c = blockIdx.x;
    __shared__ float ssum[256], ssq[256];
    float s = 0.f, q = 0.f;
    const int total = NBATCH * V;
    for (int i = threadIdx.x; i < total; i += 256) {
        int b = i / V, v = i - b * V;
        float val = x[((size_t)b * C + c) * V + v];
        s += val; q += val * val;
    }
    ssum[threadIdx.x] = s; ssq[threadIdx.x] = q;
    __syncthreads();
    for (int st = 128; st > 0; st >>= 1) {
        if (threadIdx.x < st) {
            ssum[threadIdx.x] += ssum[threadIdx.x + st];
            ssq [threadIdx.x] += ssq [threadIdx.x + st];
        }
        __syncthreads();
    }
    if (threadIdx.x == 0) {
        float inv = 1.f / (float)total;
        float m = ssum[0] * inv;
        float var = ssq[0] * inv - m * m;
        mean[c] = m;
        rstd[c] = rsqrtf(var + EPSV);
    }
}

// mode: 0 = norm only, 1 = norm+relu, 2 = norm + residual + relu
__global__ __launch_bounds__(256)
void k_bn_apply(const float* __restrict__ x, float* __restrict__ y,
                const float* __restrict__ mean, const float* __restrict__ rstd,
                const float* __restrict__ g, const float* __restrict__ bta,
                const float* __restrict__ res, int C, int V, int mode)
{
    size_t idx = (size_t)blockIdx.x * 256 + threadIdx.x;
    size_t total = (size_t)NBATCH * C * V;
    if (idx >= total) return;
    int c = (int)((idx / V) % C);
    float val = (x[idx] - mean[c]) * rstd[c] * g[c] + bta[c];
    if (mode == 2) val += res[idx];
    if (mode)      val = fmaxf(val, 0.f);
    y[idx] = val;
}

// ---------------------------------------------------------------------------
// GroupNorm (32 groups): one block per (batch, group)
// ---------------------------------------------------------------------------
__global__ __launch_bounds__(256)
void k_gn_stats(const float* __restrict__ x, int C, int V,
                float* __restrict__ mean, float* __restrict__ rstd)
{
    const int bg = blockIdx.x;
    const int b = bg >> 5, g = bg & 31;
    const int cpg = C / 32;
    __shared__ float ssum[256], ssq[256];
    float s = 0.f, q = 0.f;
    const int total = cpg * V;
    for (int i = threadIdx.x; i < total; i += 256) {
        int cl = i / V, v = i - cl * V;
        int c = g * cpg + cl;
        float val = x[((size_t)b * C + c) * V + v];
        s += val; q += val * val;
    }
    ssum[threadIdx.x] = s; ssq[threadIdx.x] = q;
    __syncthreads();
    for (int st = 128; st > 0; st >>= 1) {
        if (threadIdx.x < st) {
            ssum[threadIdx.x] += ssum[threadIdx.x + st];
            ssq [threadIdx.x] += ssq [threadIdx.x + st];
        }
        __syncthreads();
    }
    if (threadIdx.x == 0) {
        float inv = 1.f / (float)total;
        float m = ssum[0] * inv;
        float var = ssq[0] * inv - m * m;
        mean[bg] = m;
        rstd[bg] = rsqrtf(var + EPSV);
    }
}

__global__ __launch_bounds__(256)
void k_gn_apply(const float* __restrict__ x, float* __restrict__ y,
                const float* __restrict__ mean, const float* __restrict__ rstd,
                const float* __restrict__ g, const float* __restrict__ bta,
                int C, int V)
{
    size_t idx = (size_t)blockIdx.x * 256 + threadIdx.x;
    size_t total = (size_t)NBATCH * C * V;
    if (idx >= total) return;
    int c  = (int)((idx / V) % C);
    int b  = (int)(idx / ((size_t)C * V));
    int cpg = C / 32;
    int bg = b * 32 + c / cpg;
    y[idx] = (x[idx] - mean[bg]) * rstd[bg] * g[c] + bta[c];
}

// ---------------------------------------------------------------------------
// upsample(hp) + concat([skip, up(hp)]) -> out (Cs+Ch channels at V verts)
// ---------------------------------------------------------------------------
__global__ __launch_bounds__(256)
void k_upcat(const float* __restrict__ skip, const float* __restrict__ hp,
             const int* __restrict__ par, float* __restrict__ out,
             int Cs, int Ch, int V, int NVp)
{
    const int Ct = Cs + Ch;
    size_t idx = (size_t)blockIdx.x * 256 + threadIdx.x;
    size_t total = (size_t)NBATCH * Ct * V;
    if (idx >= total) return;
    int v = (int)(idx % V);
    int c = (int)((idx / V) % Ct);
    int b = (int)(idx / ((size_t)Ct * V));
    float val;
    if (c < Cs) {
        val = skip[((size_t)b * Cs + c) * V + v];
    } else {
        int ch = c - Cs;
        const float* hb = hp + ((size_t)b * Ch + ch) * NVp;
        if (v < NVp) val = hb[v];
        else {
            int r = v - NVp;
            val = 0.5f * (hb[par[2*r]] + hb[par[2*r + 1]]);
        }
    }
    out[idx] = val;
}

// ---------------------------------------------------------------------------
// Host orchestration
// ---------------------------------------------------------------------------
static inline unsigned cdiv(size_t a, size_t b) { return (unsigned)((a + b - 1) / b); }

static void conv(hipStream_t s, const float* X, int C, int Vin,
                 const float* W, const float* bias, const int* nbr, int KN,
                 int O, int Vout, float* Y, unsigned* wp, const float* zbuf)
{
    int otiles = (O + 15) / 16;
    int nc     = (C + 31) / 32;
    int totalw = otiles * KN * nc * 256;
    k_pack_w<<<cdiv(totalw, 256), 256, 0, s>>>(W, wp, O, C, KN, otiles, nc);
    dim3 grid(cdiv(Vout, 32), cdiv(otiles, 4), NBATCH);
    k_conv_wmma<<<grid, 128, 0, s>>>(X, wp, bias, nbr, zbuf, Y, C, Vin, O, Vout,
                                     KN, otiles, nc);
}

struct RBp { const float* p[16]; };
// order: 0 c1a_w 1 c1a_b 2 bn1a_g 3 bn1a_b 4 c2a_w 5 c2a_b 6 bn2a_g 7 bn2a_b
//        8 c3a_w 9 c3a_b 10 bn3a_g 11 bn3a_b 12 c1b_w 13 c1b_b 14 gn_g 15 gn_b

static void resblock(hipStream_t s, const float* x, int Cin, int Vin,
                     const RBp& P, int nk, int co, const int* nbr7, int Vout,
                     float* a, float* a2, float* h, float* mean, float* rstd,
                     unsigned* wp, const float* zbuf, float* out)
{
    // h = gnorm(conv1x1(x, conv1b)[:, :, :Vout])
    conv(s, x, Cin, Vin, P.p[12], P.p[13], nullptr, 1, co, Vout, h, wp, zbuf);
    k_gn_stats<<<NBATCH * 32, 256, 0, s>>>(h, co, Vout, mean, rstd);
    size_t th = (size_t)NBATCH * co * Vout;
    k_gn_apply<<<cdiv(th, 256), 256, 0, s>>>(h, h, mean, rstd, P.p[14], P.p[15], co, Vout);

    // a = relu(bn1a(conv1x1(x, conv1a)[:, :, :Vout]))
    conv(s, x, Cin, Vin, P.p[0], P.p[1], nullptr, 1, nk, Vout, a, wp, zbuf);
    size_t ta = (size_t)NBATCH * nk * Vout;
    k_bn_stats<<<nk, 256, 0, s>>>(a, nk, Vout, mean, rstd);
    k_bn_apply<<<cdiv(ta, 256), 256, 0, s>>>(a, a, mean, rstd, P.p[2], P.p[3], nullptr, nk, Vout, 1);

    // a2 = relu(bn2a(meshconv(a, conv2a)))
    conv(s, a, nk, Vout, P.p[4], P.p[5], nbr7, 7, nk, Vout, a2, wp, zbuf);
    k_bn_stats<<<nk, 256, 0, s>>>(a2, nk, Vout, mean, rstd);
    k_bn_apply<<<cdiv(ta, 256), 256, 0, s>>>(a2, a2, mean, rstd, P.p[6], P.p[7], nullptr, nk, Vout, 1);

    // out = relu(bn3a(conv1x1(a2, conv3a)) + h)
    conv(s, a2, nk, Vout, P.p[8], P.p[9], nullptr, 1, co, Vout, a, wp, zbuf);
    k_bn_stats<<<co, 256, 0, s>>>(a, co, Vout, mean, rstd);
    k_bn_apply<<<cdiv(th, 256), 256, 0, s>>>(a, out, mean, rstd, P.p[10], P.p[11], h, co, Vout, 2);
}

extern "C" void kernel_launch(void* const* d_in, const int* in_sizes, int n_in,
                              void* d_out, int out_size, void* d_ws, size_t ws_size,
                              hipStream_t stream)
{
    (void)in_sizes; (void)n_in; (void)out_size;
    static const int NV[6] = {12, 42, 162, 642, 2562, 10242};

    // ---- unpack inputs (setup_inputs() insertion order, recursive) ----
    int idx = 0;
    const float* x    = (const float*)d_in[idx++];
    const float* inw  = (const float*)d_in[idx++];
    const float* inb  = (const float*)d_in[idx++];
    const float* outw = (const float*)d_in[idx++];
    const float* outb = (const float*)d_in[idx++];
    RBp down[5], up[5];
    for (int i = 0; i < 5; ++i)
        for (int j = 0; j < 16; ++j) down[i].p[j] = (const float*)d_in[idx++];
    for (int i = 0; i < 5; ++i)
        for (int j = 0; j < 16; ++j) up[i].p[j] = (const float*)d_in[idx++];
    const int* nbrs[6];
    for (int L = 0; L < 6; ++L) nbrs[L] = (const int*)d_in[idx++];
    const int* pars[6]; pars[0] = nullptr;
    for (int L = 1; L < 6; ++L) pars[L] = (const int*)d_in[idx++];

    // ---- workspace layout ----
    float* ws = (float*)d_ws;
    size_t off = 0;
    auto alloc = [&](size_t n) { float* p = ws + off; off += n; return p; };
    float* skip[6];
    skip[0] = alloc((size_t)NBATCH * 64   * 10242);
    skip[1] = alloc((size_t)NBATCH * 128  * 2562);
    skip[2] = alloc((size_t)NBATCH * 256  * 642);
    skip[3] = alloc((size_t)NBATCH * 512  * 162);
    skip[4] = alloc((size_t)NBATCH * 1024 * 42);
    skip[5] = alloc((size_t)NBATCH * 1024 * 12);
    float* buf0 = alloc((size_t)NBATCH * 128 * 10242);  // concat / x
    float* buf1 = alloc((size_t)NBATCH * 64  * 10242);  // a / a3
    float* buf2 = alloc((size_t)NBATCH * 64  * 10242);  // a2 / resblock out
    float* buf3 = alloc((size_t)NBATCH * 64  * 10242);  // h branch
    float* mean = alloc(1024);
    float* rstd = alloc(1024);
    float* zbuf = alloc(64);                             // zero pad source
    unsigned* wp = (unsigned*)alloc(64 * 7 * 32 * 256);  // packed weights (max layer)
    if (off * sizeof(float) > ws_size) return;           // workspace too small

    k_zero<<<1, 64, 0, stream>>>(zbuf, 64);

    // ---- in_conv: meshconv 8 -> 64 at level 5 ----
    conv(stream, x, 8, 10242, inw, inb, nbrs[5], 7, 64, 10242, skip[0], wp, zbuf);

    // ---- down path ----
    static const int dci[5]  = {64, 128, 256, 512, 1024};
    static const int dco[5]  = {128, 256, 512, 1024, 1024};
    static const int dlvl[5] = {4, 3, 2, 1, 0};
    for (int i = 0; i < 5; ++i) {
        int lvl = dlvl[i];
        resblock(stream, skip[i], dci[i], NV[lvl + 1], down[i], dci[i], dco[i],
                 nbrs[lvl], NV[lvl], buf1, buf2, buf3, mean, rstd, wp, zbuf, skip[i + 1]);
    }

    // ---- up path ----
    static const int upCs[5] = {1024, 512, 256, 128, 64};
    static const int upCo[5] = {512, 256, 128, 64, 64};
    const float* hcur = skip[5];
    int Chh = 1024;
    for (int i = 0; i < 5; ++i) {
        int lvl = i + 1;
        int V = NV[lvl], NVp = NV[lvl - 1];
        int Ct = upCs[i] + Chh;
        size_t tc = (size_t)NBATCH * Ct * V;
        k_upcat<<<cdiv(tc, 256), 256, 0, stream>>>(skip[4 - i], hcur, pars[lvl],
                                                   buf0, upCs[i], Chh, V, NVp);
        resblock(stream, buf0, Ct, V, up[i], upCo[i], upCo[i],
                 nbrs[lvl], V, buf1, buf2, buf3, mean, rstd, wp, zbuf, buf2);
        hcur = buf2;
        Chh = upCo[i];
    }

    // ---- out_conv: meshconv 64 -> 8 at level 5 ----
    conv(stream, hcur, 64, 10242, outw, outb, nbrs[5], 7, 8, 10242, (float*)d_out, wp, zbuf);
}
